// NetVLAD_7413113553696
// MI455X (gfx1250) — matrix operand order repaired and checked by reading, not compiled, not run on hardware
//
#include <hip/hip_runtime.h>

typedef __attribute__((ext_vector_type(16))) __bf16 v16bf;
typedef __attribute__((ext_vector_type(8)))  float  v8f;

#define N_IMG   32
#define C_DIM   128
#define S_DIM   4096
#define K_DIM   64
#define T_PIX   128
#define SPLIT   8
#define NCHUNK  (S_DIM / (SPLIT * T_PIX))   // 4

// LDS pitches (keep fragment starts 16B-aligned, skew banks)
#define PW 136   // pitch in halves for bf16 tiles (272 B/row)
#define PL 132   // pitch in floats for logits      (528 B/row)
#define PF 132   // pitch in floats for raw x tile

// LDS byte layout
#define OFF_W    0
#define SZ_W     (K_DIM * PW * 2)              // 17408
#define OFF_XCS  (OFF_W + SZ_W)                // 17408
#define SZ_XCS   (C_DIM * PW * 2)              // 34816
#define OFF_XSC  (OFF_XCS + SZ_XCS)            // 52224
#define SZ_XSC   (T_PIX * PW * 2)              // 34816
#define OFF_SCR  (OFF_XSC + SZ_XSC)            // 87040  (union: Xf | Lg+Abf)
#define SZ_XF    (C_DIM * PF * 4)              // 67584
#define OFF_LG   OFF_SCR
#define SZ_LG    (K_DIM * PL * 4)              // 33792
#define OFF_ABF  (OFF_LG + SZ_LG)              // 120832
#define SZ_ABF   (K_DIM * PW * 2)              // 17408  (ends 138240 < OFF_RED)
#define OFF_RED  (OFF_SCR + SZ_XF)             // 154624
#define SZ_RED   (512 * 4)
#define OFF_CB   (OFF_RED + SZ_RED)            // 156672
#define SZ_CB    (64 * 4)
#define SMEM_BYTES (OFF_CB + SZ_CB)            // 156928 B (~153 KB)

union Frag {
  uint4 q[2];
  v16bf v;
};

__device__ __forceinline__ unsigned short f2bf(float f) {
  unsigned int u = __float_as_uint(f);
  u += 0x7FFFu + ((u >> 16) & 1u);   // RNE
  return (unsigned short)(u >> 16);
}
__device__ __forceinline__ float bf2f(unsigned short h) {
  return __uint_as_float(((unsigned int)h) << 16);
}

// -------- Kernel 1: fused normalize + GEMM1 + softmax + GEMM2 (partial vlad) --------
__global__ __launch_bounds__(256) void netvlad_main(
    const float* __restrict__ x, const float* __restrict__ conv_w,
    const float* __restrict__ conv_b, float* __restrict__ vladP,
    float* __restrict__ asumP)
{
  __shared__ unsigned char smem[SMEM_BYTES] __attribute__((aligned(128)));
  unsigned short* Wbf = (unsigned short*)(smem + OFF_W);
  unsigned short* Xcs = (unsigned short*)(smem + OFF_XCS);  // x_norm[c][s] bf16
  unsigned short* Xsc = (unsigned short*)(smem + OFF_XSC);  // x_norm[s][c] bf16
  float*          Xf  = (float*)(smem + OFF_SCR);           // raw x[c][s] f32
  float*          Lg  = (float*)(smem + OFF_LG);            // logits[k][s] f32
  unsigned short* Abf = (unsigned short*)(smem + OFF_ABF);  // assign[k][s] bf16
  float*          red = (float*)(smem + OFF_RED);
  float*          cbS = (float*)(smem + OFF_CB);

  const int tid   = threadIdx.x;
  const int lane  = tid & 31;
  const int wave  = tid >> 5;
  const int blk   = blockIdx.x;
  const int n     = blk / SPLIT;
  const int split = blk % SPLIT;
  const int sbase = split * (S_DIM / SPLIT);

  // stage conv weights (bf16) + bias into LDS
  for (int i = tid; i < K_DIM * C_DIM; i += 256)
    Wbf[(i >> 7) * PW + (i & 127)] = f2bf(conv_w[i]);
  if (tid < K_DIM) cbS[tid] = conv_b[tid];

  const int mt = wave & 3;       // k-row tile (0..3)
  const int qb = wave >> 2;      // column-tile group (0..1)
  const int hh = lane >> 4;      // lane half
  const int nn = lane & 15;

  v8f acc[4];
  #pragma unroll
  for (int j = 0; j < 4; ++j)
    #pragma unroll
    for (int r = 0; r < 8; ++r) acc[j][r] = 0.0f;
  float asum_acc = 0.0f;

  const int sl  = tid & 127;     // pixel within chunk
  const int ch0 = tid >> 7;      // channel parity (0/1)
  const float* xbase = x + (size_t)n * C_DIM * S_DIM + sbase + sl;
  const float* xtile0 = x + (size_t)n * C_DIM * S_DIM + sbase;   // un-offset tile base

  for (int chunk = 0; chunk < NCHUNK; ++chunk) {
    __syncthreads();  // previous iteration fully consumed LDS
    // ---- load x tile + per-pixel sum of squares ----
    const float* xp = xbase + chunk * T_PIX;
    float ssq = 0.0f;
    #pragma unroll 4
    for (int i = 0; i < C_DIM / 2; ++i) {
      const int c = ch0 + 2 * i;
      float v = xp[(size_t)c * S_DIM];
      Xf[c * PF + sl] = v;
      ssq += v * v;
    }
    red[tid] = ssq;
    // ---- prefetch next chunk's x tile (overlaps with GEMM/softmax phases) ----
    // 128 rows x 512 B = 512 cachelines of 128 B; emits global_prefetch_b8
    if (chunk + 1 < NCHUNK) {
      const float* xnext = xtile0 + (chunk + 1) * T_PIX;
      #pragma unroll
      for (int i = 0; i < 2; ++i) {
        const int line = tid + 256 * i;            // 0..511
        const int c    = line >> 2;                // row
        const int off  = (line & 3) * 32;          // 128 B steps in floats
        __builtin_prefetch(xnext + (size_t)c * S_DIM + off, 0, 3);
      }
    }
    __syncthreads();
    if (tid < T_PIX) {
      float tot = red[tid] + red[tid + 128];
      red[tid] = 1.0f / fmaxf(sqrtf(tot), 1e-12f);   // matches F.normalize eps
    }
    __syncthreads();
    // ---- normalize, emit bf16 in both layouts ----
    {
      const float invn = red[sl];
      #pragma unroll 4
      for (int i = 0; i < C_DIM / 2; ++i) {
        const int c = ch0 + 2 * i;
        unsigned short h = f2bf(Xf[c * PF + sl] * invn);
        Xcs[c * PW + sl] = h;
        Xsc[sl * PW + c] = h;
      }
    }
    __syncthreads();
    // ---- GEMM1: logits[64,T] = W[64,C] * Xn[C,T] ----
    #pragma unroll
    for (int j = 0; j < 4; ++j) {
      const int st = qb * 4 + j;
      v8f lg = {0.f,0.f,0.f,0.f,0.f,0.f,0.f,0.f};
      #pragma unroll
      for (int cb = 0; cb < 4; ++cb) {
        Frag a, b;
        const unsigned short* ar = Wbf + (mt * 16 + nn) * PW + cb * 32 + 8 * hh;
        a.q[0] = *(const uint4*)(ar);        // K = 8h+0..7
        a.q[1] = *(const uint4*)(ar + 16);   // K = 16+8h+0..7
        const unsigned short* br = Xsc + (st * 16 + nn) * PW + cb * 32 + 8 * hh;
        b.q[0] = *(const uint4*)(br);        // K = 8h+0..7
        b.q[1] = *(const uint4*)(br + 16);   // K = 16+8h+0..7
        lg = __builtin_amdgcn_wmma_f32_16x16x32_bf16(false, a.v, false, b.v,
                                                     (short)0, lg, false, false);
      }
      #pragma unroll
      for (int r = 0; r < 8; ++r)
        Lg[(mt * 16 + hh * 8 + r) * PL + st * 16 + nn] = lg[r];
    }
    __syncthreads();
    // ---- softmax over K per pixel (+bias) ----
    if (tid < T_PIX) {
      float m = -3.0e38f;
      #pragma unroll 8
      for (int k = 0; k < K_DIM; ++k)
        m = fmaxf(m, Lg[k * PL + tid] + cbS[k]);
      float sum = 0.0f;
      #pragma unroll 8
      for (int k = 0; k < K_DIM; ++k) {
        float e = __expf(Lg[k * PL + tid] + cbS[k] - m);
        Lg[k * PL + tid] = e;
        sum += e;
      }
      const float rs = 1.0f / sum;
      #pragma unroll 8
      for (int k = 0; k < K_DIM; ++k)
        Abf[k * PW + tid] = f2bf(Lg[k * PL + tid] * rs);
    }
    __syncthreads();
    // ---- asum[k] += sum_s a[k][s] ----
    {
      const int kk = tid >> 2, q = tid & 3;
      float part = 0.0f;
      #pragma unroll 8
      for (int j = 0; j < 32; ++j)
        part += bf2f(Abf[kk * PW + q * 32 + j]);
      red[tid] = part;
      __syncthreads();
      if (tid < K_DIM)
        asum_acc += red[4*tid] + red[4*tid+1] + red[4*tid+2] + red[4*tid+3];
    }
    __syncthreads();
    // ---- GEMM2: vlad[64,C] += A[64,T] * Xn^T[T,C] ----
    #pragma unroll
    for (int j = 0; j < 4; ++j) {
      const int ct = qb * 4 + j;
      #pragma unroll
      for (int tb = 0; tb < 4; ++tb) {
        Frag a, b;
        const unsigned short* ar = Abf + (mt * 16 + nn) * PW + tb * 32 + 8 * hh;
        a.q[0] = *(const uint4*)(ar);
        a.q[1] = *(const uint4*)(ar + 16);
        const unsigned short* br = Xcs + (ct * 16 + nn) * PW + tb * 32 + 8 * hh;
        b.q[0] = *(const uint4*)(br);
        b.q[1] = *(const uint4*)(br + 16);
        acc[j] = __builtin_amdgcn_wmma_f32_16x16x32_bf16(false, a.v, false, b.v,
                                                         (short)0, acc[j], false, false);
      }
    }
  }
  __syncthreads();
  // ---- spill partials to workspace ----
  if (tid < K_DIM) asumP[(size_t)blk * K_DIM + tid] = asum_acc;
  float* vp = vladP + (size_t)blk * K_DIM * C_DIM;
  #pragma unroll
  for (int j = 0; j < 4; ++j) {
    const int ct = qb * 4 + j;
    #pragma unroll
    for (int r = 0; r < 8; ++r)
      vp[(mt * 16 + hh * 8 + r) * C_DIM + ct * 16 + nn] = acc[j][r];
  }
}

// -------- Kernel 2: reduce splits, -asum*centroids, intra-norm, fc --------
__global__ __launch_bounds__(128) void netvlad_finish(
    const float* __restrict__ vladP, const float* __restrict__ asumP,
    const float* __restrict__ centroids, const float* __restrict__ fcw,
    float* __restrict__ out)
{
  __shared__ float sred[128];
  const int n = blockIdx.x;
  const int c = threadIdx.x;
  float o = 0.0f;
  for (int k = 0; k < K_DIM; ++k) {
    float asum = 0.0f;
    #pragma unroll
    for (int p = 0; p < SPLIT; ++p)
      asum += asumP[(size_t)(n * SPLIT + p) * K_DIM + k];
    float v = 0.0f;
    #pragma unroll
    for (int p = 0; p < SPLIT; ++p)
      v += vladP[((size_t)(n * SPLIT + p) * K_DIM + k) * C_DIM + c];
    v -= asum * centroids[k * C_DIM + c];
    sred[c] = v * v;
    __syncthreads();
    for (int off = 64; off > 0; off >>= 1) {
      if (c < off) sred[c] += sred[c + off];
      __syncthreads();
    }
    const float scale = fcw[k] / fmaxf(sqrtf(sred[0]), 1e-12f);
    __syncthreads();
    o += v * scale;
  }
  out[n * C_DIM + c] = o;
}

extern "C" void kernel_launch(void* const* d_in, const int* in_sizes, int n_in,
                              void* d_out, int out_size, void* d_ws, size_t ws_size,
                              hipStream_t stream) {
  (void)in_sizes; (void)n_in; (void)out_size; (void)ws_size;
  const float* x         = (const float*)d_in[0];
  const float* conv_w    = (const float*)d_in[1];
  const float* conv_b    = (const float*)d_in[2];
  const float* centroids = (const float*)d_in[3];
  const float* fc_w      = (const float*)d_in[4];
  float* out = (float*)d_out;

  float* vladP = (float*)d_ws;                                   // [N*SPLIT][K][C]
  float* asumP = vladP + (size_t)N_IMG * SPLIT * K_DIM * C_DIM;  // [N*SPLIT][K]

  netvlad_main<<<N_IMG * SPLIT, 256, 0, stream>>>(x, conv_w, conv_b, vladP, asumP);
  netvlad_finish<<<N_IMG, 128, 0, stream>>>(vladP, asumP, centroids, fc_w, out);
}
